// PlanarPtychographyForwardModel_89197880803925
// MI455X (gfx1250) — compile-verified, matrix operand order checked by reading, not run on hardware
//
#include <hip/hip_runtime.h>
#include <math.h>

// ---------------- problem constants ----------------
#define NPOS     512
#define BATCH    32
#define NOPR     4
#define NMODES   8
#define HOBJ     1024
#define WOBJ     1024
#define HP       256
#define WP       256
#define NSLICES  4
#define NIMG     (BATCH * NMODES)        // 256 complex images of 256x256
#define NPIX     (HP * WP)               // 65536
#define PI_F     3.14159265358979323846f

typedef __attribute__((ext_vector_type(16))) _Float16 v16h;
typedef __attribute__((ext_vector_type(8)))  float    v8f;

union V16H { v16h v; _Float16 e[16]; };
union V8F  { v8f  v; float    e[8];  };
union H2   { unsigned int u; _Float16 h[2]; };   // packed (re, im) f16 pair

__device__ inline unsigned int pack_h2(float re, float im) {
  H2 a; a.h[0] = (_Float16)re; a.h[1] = (_Float16)im; return a.u;
}
__device__ inline float2 unpack_h2(unsigned int u) {
  H2 a; a.u = u; float2 v; v.x = (float)a.h[0]; v.y = (float)a.h[1]; return v;
}

// =====================================================================
// In-LDS WMMA radix-16 FFT pass over a 256x256 complex-f16 tile.
// One 1-D 256-pt FFT per wave per line; 8 waves cover 256 lines.
// 256-pt DFT = (DFT16 x) -> twiddle W256^{n2 k1} -> (x DFT16); each
// complex 16x16x16 matmul is ONE v_wmma_f32_16x16x32_f16 per component
// via K-concatenation: Cr=[Ar|Ai][[Br],[-Bi]], Ci=[Ar|Ai][[Bi],[Br]].
// col=0: lines are rows (tile[line][i]); col=1: lines are columns.
// sgn = -1 forward, +1 inverse; scale applied on output.
// Caller provides __syncthreads() fencing between passes.
// =====================================================================
__device__ void fft_pass_lds(unsigned int (*tile)[WP + 1],
                             float (*stR)[16][17], float (*stI)[16][17],
                             int col, float sgn, float scale) {
  const int lane = threadIdx.x & 31;
  const int wave = threadIdx.x >> 5;
  const int g    = lane >> 4;   // lane group (K half / row half)
  const int ln   = lane & 15;   // row m (A) / col n (B, C)

  const float W16TH = 2.0f * PI_F / 16.0f;

  // constant fragments: stage-1 A = [W16r | W16i], stage-2 B pair
  V16H aW, b2re, b2im;
#pragma unroll
  for (int h = 0; h < 16; ++h) {
    const int v = h >> 1, c = h & 1;
    const int k  = ((v < 4) ? 0 : 16) + ((v & 3) << 1) + (g << 3) + c;
    const int kk = (k < 16) ? k : (k - 16);
    const float th = W16TH * (float)((ln * kk) & 15);
    const float cv = __cosf(th), sv = __sinf(th);
    aW.e[h] = (_Float16)((k < 16) ? cv : sgn * sv);
    if (k < 16) { b2re.e[h] = (_Float16)cv;          b2im.e[h] = (_Float16)(sgn * sv); }
    else        { b2re.e[h] = (_Float16)(-sgn * sv); b2im.e[h] = (_Float16)cv; }
  }

  // twiddles W256^{sgn*k1*n2}: line-invariant, hoisted
  float twr[8], twi[8];
#pragma unroll
  for (int r = 0; r < 8; ++r) {
    const float th = (2.0f * PI_F / 256.0f) * (float)((r + (g << 3)) * ln);
    twr[r] = __cosf(th); twi[r] = sgn * __sinf(th);
  }

  for (int line = wave; line < 256; line += 8) {
    // stage-1 B fragments: X[n1][n2] = x[16*n1 + n2], data already f16
    H2 xv[8];
#pragma unroll
    for (int j = 0; j < 8; ++j) {
      const int i = 16 * ((g << 3) + j) + ln;
      xv[j].u = col ? tile[i][line] : tile[line][i];
    }
    V16H b1re, b1im;
#pragma unroll
    for (int h = 0; h < 16; ++h) {
      const int v = h >> 1, c = h & 1;
      const int j = ((v & 3) << 1) + c;
      if (v < 4) { b1re.e[h] = xv[j].h[0];  b1im.e[h] = xv[j].h[1]; }
      else       { b1re.e[h] = -xv[j].h[1]; b1im.e[h] = xv[j].h[0]; }
    }

    V8F accRe, accIm;
#pragma unroll
    for (int r = 0; r < 8; ++r) { accRe.e[r] = 0.f; accIm.e[r] = 0.f; }
    accRe.v = __builtin_amdgcn_wmma_f32_16x16x32_f16(false, aW.v, false, b1re.v,
                                                     (short)0, accRe.v, false, false);
    accIm.v = __builtin_amdgcn_wmma_f32_16x16x32_f16(false, aW.v, false, b1im.v,
                                                     (short)0, accIm.v, false, false);

    // twiddle, stash T into per-wave LDS staging (DS in-order within a wave)
#pragma unroll
    for (int r = 0; r < 8; ++r) {
      const int k1 = r + (g << 3);
      const float re = accRe.e[r], im = accIm.e[r];
      stR[wave][k1][ln] = re * twr[r] - im * twi[r];
      stI[wave][k1][ln] = re * twi[r] + im * twr[r];
    }

    // stage-2 A fragment (data): [Tr | Ti], row m=ln
    V16H a2;
#pragma unroll
    for (int h = 0; h < 16; ++h) {
      const int v = h >> 1, c = h & 1;
      const int k = ((v < 4) ? 0 : 16) + ((v & 3) << 1) + (g << 3) + c;
      a2.e[h] = (_Float16)((k < 16) ? stR[wave][ln][k] : stI[wave][ln][k - 16]);
    }

    V8F oRe, oIm;
#pragma unroll
    for (int r = 0; r < 8; ++r) { oRe.e[r] = 0.f; oIm.e[r] = 0.f; }
    oRe.v = __builtin_amdgcn_wmma_f32_16x16x32_f16(false, a2.v, false, b2re.v,
                                                   (short)0, oRe.v, false, false);
    oIm.v = __builtin_amdgcn_wmma_f32_16x16x32_f16(false, a2.v, false, b2im.v,
                                                   (short)0, oIm.v, false, false);

    // scatter: out index k = k1 + 16*k2 (k1 = C row, k2 = C col)
#pragma unroll
    for (int r = 0; r < 8; ++r) {
      const int k = (r + (g << 3)) + 16 * ln;
      const unsigned int u = pack_h2(oRe.e[r] * scale, oIm.e[r] * scale);
      if (col) tile[k][line] = u; else tile[line][k] = u;
    }
  }
}

// =====================================================================
// Angular-spectrum transfer function (f64 trig: dz*kz ~ 1e5 rad)
// =====================================================================
__global__ __launch_bounds__(256) void k_tf_build(float2* __restrict__ tf) {
  const int p = blockIdx.x * 256 + threadIdx.x;
  const int y = p >> 8, x = p & 255;
  const double lam = 1.24e-10, pix = 1.0e-8, dz = 2.0e-6;
  const double fy = (double)((y < 128) ? y : y - 256) / (256.0 * pix);
  const double fx = (double)((x < 128) ? x : x - 256) / (256.0 * pix);
  const double arg = 1.0 - lam * lam * (fx * fx + fy * fy);
  float2 o; o.x = 0.f; o.y = 0.f;
  if (arg > 0.0) {
    const double kz = 2.0 * M_PI / lam * sqrt(arg);
    double s, c; sincos(dz * kz, &s, &c);
    o.x = (float)c; o.y = (float)s;
  }
  tf[p] = o;
}

// =====================================================================
// Fused forward model: one block per (batch, mode) image; psi lives in
// LDS (f16 complex, 257 KB) for the entire multislice pipeline. All
// global inputs (~50 MB) are L2-resident on MI455X (192 MB L2).
// =====================================================================
__global__ __launch_bounds__(256) void k_forward_fused(
    const float2* __restrict__ obj,
    const float2* __restrict__ probe,
    const float*  __restrict__ oprw,
    const float*  __restrict__ positions,
    const int*    __restrict__ idx,
    const float2* __restrict__ tf,
    float*        __restrict__ inten) {

  __shared__ unsigned int tile[HP][WP + 1];    // 263,168 B packed f16 complex
  __shared__ float stR[8][16][17];             //  8,704 B per-wave transpose tile
  __shared__ float stI[8][16][17];             //  8,704 B

  const int im  = blockIdx.x;
  const int b   = im >> 3, m = im & 7;
  const int tid = threadIdx.x;

  const int   id = idx[b];
  const float py = positions[id * 2 + 0], px = positions[id * 2 + 1];
  const float ry = rintf(py), rx = rintf(px);       // ties-to-even, as jnp.round
  const int   iy = (int)ry,  ix = (int)rx;
  const float fr = py - ry,  fc = px - rx;

  // ---- assemble uprobe into LDS ----
  if (m == 0) {
    float w[NOPR];
#pragma unroll
    for (int k = 0; k < NOPR; ++k) w[k] = oprw[id * NOPR + k];
    for (int p = tid; p < NPIX; p += 256) {
      float ar = 0.f, ai = 0.f;
#pragma unroll
      for (int k = 0; k < NOPR; ++k) {
        const float2 pv = probe[(size_t)(k * NMODES) * NPIX + p];
        ar += w[k] * pv.x; ai += w[k] * pv.y;
      }
      tile[p >> 8][p & 255] = pack_h2(ar, ai);
    }
  } else {
    for (int p = tid; p < NPIX; p += 256) {
      const float2 pv = probe[(size_t)m * NPIX + p];
      tile[p >> 8][p & 255] = pack_h2(pv.x, pv.y);
    }
  }
  __syncthreads();

  // ---- subpixel Fourier shift: FFT2 -> ramp -> IFFT2 ----
  fft_pass_lds(tile, stR, stI, 0, -1.f, 1.f);  __syncthreads();
  fft_pass_lds(tile, stR, stI, 1, -1.f, 1.f);  __syncthreads();
  for (int p = tid; p < NPIX; p += 256) {
    const int y = p >> 8, x = p & 255;
    const float fy = (float)((y < 128) ? y : y - 256) * (1.0f / 256.0f);
    const float fx = (float)((x < 128) ? x : x - 256) * (1.0f / 256.0f);
    const float th = -2.0f * PI_F * (fr * fy + fc * fx);
    const float c = __cosf(th), s = __sinf(th);
    const float2 v = unpack_h2(tile[y][x]);
    tile[y][x] = pack_h2(v.x * c - v.y * s, v.x * s + v.y * c);
  }
  __syncthreads();
  fft_pass_lds(tile, stR, stI, 0, 1.f, 1.f / 256.f);  __syncthreads();
  fft_pass_lds(tile, stR, stI, 1, 1.f, 1.f / 256.f);  __syncthreads();

  // ---- multislice propagation ----
  for (int s = 0; s < NSLICES; ++s) {
    for (int p = tid; p < NPIX; p += 256) {
      const int y = p >> 8, x = p & 255;
      const float2 ov = obj[((size_t)s * HOBJ + (iy + y)) * WOBJ + (ix + x)];
      const float2 v  = unpack_h2(tile[y][x]);
      float orr = v.x * ov.x - v.y * ov.y;
      float oii = v.x * ov.y + v.y * ov.x;
      // fold ifftshift/fftshift of the final far-field transform into a
      // (-1)^(x+y) checkerboard on the last slice (phase dies in |.|^2)
      if (s == NSLICES - 1 && (((x + y) & 1) != 0)) { orr = -orr; oii = -oii; }
      tile[y][x] = pack_h2(orr, oii);
    }
    __syncthreads();
    if (s < NSLICES - 1) {
      fft_pass_lds(tile, stR, stI, 0, -1.f, 1.f);  __syncthreads();
      fft_pass_lds(tile, stR, stI, 1, -1.f, 1.f);  __syncthreads();
      for (int p = tid; p < NPIX; p += 256) {
        const int y = p >> 8, x = p & 255;
        const float2 h = tf[p];
        const float2 v = unpack_h2(tile[y][x]);
        tile[y][x] = pack_h2(v.x * h.x - v.y * h.y, v.x * h.y + v.y * h.x);
      }
      __syncthreads();
      fft_pass_lds(tile, stR, stI, 0, 1.f, 1.f / 256.f);  __syncthreads();
      fft_pass_lds(tile, stR, stI, 1, 1.f, 1.f / 256.f);  __syncthreads();
    }
  }

  // ---- far-field FFT2, norm='ortho' (1/16 per 1-D pass) ----
  fft_pass_lds(tile, stR, stI, 0, -1.f, 1.f / 16.f);  __syncthreads();
  fft_pass_lds(tile, stR, stI, 1, -1.f, 1.f / 16.f);  __syncthreads();

  // ---- per-mode intensity plane ----
  for (int p = tid; p < NPIX; p += 256) {
    const float2 v = unpack_h2(tile[p >> 8][p & 255]);
    inten[(size_t)im * NPIX + p] = v.x * v.x + v.y * v.y;
  }
}

// =====================================================================
// deterministic incoherent mode sum (no float atomics -> bit-stable replays)
// =====================================================================
__global__ __launch_bounds__(256) void k_intensity(const float* __restrict__ inten,
                                                   float* __restrict__ out) {
  const int t = blockIdx.x * 256 + threadIdx.x;   // BATCH * NPIX
  const int p = t & (NPIX - 1);
  const int b = t >> 16;
  float acc = 0.f;
#pragma unroll
  for (int m = 0; m < NMODES; ++m)
    acc += inten[((size_t)(b * NMODES + m)) * NPIX + p];
  out[(size_t)t] = acc;
}

// =====================================================================
extern "C" void kernel_launch(void* const* d_in, const int* in_sizes, int n_in,
                              void* d_out, int out_size, void* d_ws, size_t ws_size,
                              hipStream_t stream) {
  const float2* obj       = (const float2*)d_in[0];   // (S,H,W,2) f32
  const float2* probe     = (const float2*)d_in[1];   // (K,M,h,w,2) f32
  const float*  oprw      = (const float*)d_in[2];    // (NPOS,K)
  const float*  positions = (const float*)d_in[3];    // (NPOS,2)
  const int*    idx       = (const int*)d_in[4];      // (B,)
  float*        out       = (float*)d_out;

  float*  inten = (float*)d_ws;                        // NIMG*NPIX f32 (64 MB)
  float2* tf    = (float2*)(inten + (size_t)NIMG * NPIX);  // NPIX complex (512 KB)

  const dim3 B(256);
  k_tf_build      <<<NPIX / 256, B, 0, stream>>>(tf);
  k_forward_fused <<<NIMG,       B, 0, stream>>>(obj, probe, oprw, positions,
                                                 idx, tf, inten);
  k_intensity     <<<(BATCH * NPIX) / 256, B, 0, stream>>>(inten, out);
}
